// RelationalGraphLayer_17033840296196
// MI455X (gfx1250) — matrix operand order repaired and checked by reading, not compiled, not run on hardware
//
#include <hip/hip_runtime.h>
#include <hip/hip_bf16.h>

typedef __attribute__((ext_vector_type(16))) _Float16 v16h;
typedef __attribute__((ext_vector_type(8)))  float    v8f;

#define N_NODES 50000
#define N_EDGES 800000
#define DIMD 128
#define NREL 4
#define HID 64
#define WAVES 8
// edges are bucketed by relation; buckets padded to multiples of 32 edges so a
// 32-edge "pair" (2 WMMA tiles) never straddles a relation boundary.
#define PPAIRS (N_EDGES / 32 + NREL)
#define PERM_N (N_EDGES + NREL * 32)
// meta int indices
#define M_CNT 0   // counts[4]
#define M_OFF 4   // offsets[5] (32-padded, offsets[4] = padded total)
#define M_CUR 12  // cursors[4]

// ---------------- WMMA + layout helpers ----------------

static __device__ __forceinline__ v8f wmma_f16(v16h a, v16h b, v8f c) {
  return __builtin_amdgcn_wmma_f32_16x16x32_f16(false, a, false, b, (short)0, c,
                                                false, false);
}

// A-matrix 16x32 f16 layout: lane L holds row M = L&15; halves t=0..7 are
// K = 8*(L>>4)+t, halves t=8..15 are K = 16 + 8*(L>>4) + (t-8).
static __device__ __forceinline__ v16h load_a_f32(const float* __restrict__ p) {
  v16h a;
#pragma unroll
  for (int t = 0; t < 8; ++t) a[t] = (_Float16)p[t];
#pragma unroll
  for (int t = 0; t < 8; ++t) a[8 + t] = (_Float16)p[16 + t];
  return a;
}

static __device__ __forceinline__ v16h load_a_f32_relu_scale(const float* __restrict__ p, float s) {
  v16h a;
#pragma unroll
  for (int t = 0; t < 8; ++t) a[t] = (_Float16)(fmaxf(p[t], 0.f) * s);
#pragma unroll
  for (int t = 0; t < 8; ++t) a[8 + t] = (_Float16)(fmaxf(p[16 + t], 0.f) * s);
  return a;
}

// Pre-swizzled B tile: 16 contiguous halves per lane (2x b128 loads, WGP$-hot).
static __device__ __forceinline__ v16h load_b_tile(const _Float16* __restrict__ base,
                                                   int tileIdx, int lane) {
  return *(const v16h*)(base + ((size_t)tileIdx * 32 + lane) * 16);
}

static __device__ __forceinline__ v8f splat8(float x) {
  v8f c;
#pragma unroll
  for (int v = 0; v < 8; ++v) c[v] = x;
  return c;
}

// C/D layout: lane L, VGPR v holds element (M = v + 8*(L>>4), N = colBase + (L&15)).
static __device__ __forceinline__ void store_c_relu_lds(_Float16* hb, v8f c, int colBase,
                                                        int laneLo, int laneHi) {
#pragma unroll
  for (int v = 0; v < 8; ++v)
    hb[(v + 8 * laneHi) * HID + colBase + laneLo] = (_Float16)fmaxf(c[v], 0.f);
}

static __device__ __forceinline__ v16h load_a_lds(const _Float16* hb, int cc,
                                                  int laneLo, int laneHi) {
  const _Float16* p = hb + laneLo * HID + cc * 32 + 8 * laneHi;
  v16h a;
#pragma unroll
  for (int t = 0; t < 8; ++t) a[t] = p[t];
#pragma unroll
  for (int t = 0; t < 8; ++t) a[8 + t] = p[16 + t];
  return a;
}

// ---------------- Prep kernels ----------------

__global__ void swizzle_w_kernel(const float* __restrict__ src, _Float16* __restrict__ dst,
                                 int K, int Ncols) {
  int total = K * Ncols;
  for (int i = blockIdx.x * blockDim.x + threadIdx.x; i < total;
       i += gridDim.x * blockDim.x) {
    int t = i & 15;
    int lane = (i >> 4) & 31;
    int tileIdx = i >> 9;
    int nTiles = Ncols >> 4;
    int kc = tileIdx / nTiles;
    int nc = tileIdx - kc * nTiles;
    int k = kc * 32 + 8 * (lane >> 4) + (t & 7) + 16 * (t >> 3);
    int n = nc * 16 + (lane & 15);
    dst[i] = (_Float16)src[(size_t)k * Ncols + n];
  }
}

__global__ void hasin_kernel(const int* __restrict__ dst, float* __restrict__ hasin, int E) {
  int i = blockIdx.x * blockDim.x + threadIdx.x;
  if (i < E) hasin[dst[i]] = 1.0f;
}

// relation histogram (LDS-aggregated: 4 global atomics per block)
__global__ void hist_kernel(const int* __restrict__ et, int* __restrict__ meta, int E) {
  __shared__ int lcnt[NREL];
  if (threadIdx.x < NREL) lcnt[threadIdx.x] = 0;
  __syncthreads();
  int i = blockIdx.x * blockDim.x + threadIdx.x;
  if (i < E) atomicAdd(&lcnt[et[i]], 1);
  __syncthreads();
  if (threadIdx.x < NREL) atomicAdd(&meta[M_CNT + threadIdx.x], lcnt[threadIdx.x]);
}

// tiny exclusive scan with 32-padding; init cursors
__global__ void scan_kernel(int* __restrict__ meta) {
  int off = 0;
#pragma unroll
  for (int r = 0; r < NREL; ++r) {
    meta[M_OFF + r] = off;
    meta[M_CUR + r] = off;
    int c = meta[M_CNT + r];
    off += (c + 31) & ~31;
  }
  meta[M_OFF + NREL] = off;
}

// bucket placement (block-aggregated: 4 global atomics per block)
__global__ void place_kernel(const int* __restrict__ et, int* __restrict__ meta,
                             int* __restrict__ perm, int E) {
  __shared__ int lcnt[NREL];
  __shared__ int lbase[NREL];
  if (threadIdx.x < NREL) lcnt[threadIdx.x] = 0;
  __syncthreads();
  int i = blockIdx.x * blockDim.x + threadIdx.x;
  int r = -1, lpos = 0;
  if (i < E) {
    r = et[i];
    lpos = atomicAdd(&lcnt[r], 1);
  }
  __syncthreads();
  if (threadIdx.x < NREL)
    lbase[threadIdx.x] = atomicAdd(&meta[M_CUR + threadIdx.x], lcnt[threadIdx.x]);
  __syncthreads();
  if (i < E) perm[lbase[r] + lpos] = i;
}

// ---------------- Edge message kernel ----------------
// One wave per relation-pure PAIR of 16-edge tiles (32 edges). Each preloaded
// B-tile group is reused by two independent A tiles -> halves B-operand
// traffic per edge and gives two independent WMMA chains to hide B latency.

__global__ void __launch_bounds__(256) edge_msg_kernel(
    const float* __restrict__ nf, const int* __restrict__ src,
    const int* __restrict__ dst, const int* __restrict__ meta,
    const int* __restrict__ perm,
    const float* __restrict__ b1, const float* __restrict__ b2,
    const float* __restrict__ b3,
    const _Float16* __restrict__ B1s, const _Float16* __restrict__ B2s,
    const _Float16* __restrict__ B3s,
    float* __restrict__ agg) {
  __shared__ _Float16 hbuf[WAVES][2][16 * HID];
  const int w = threadIdx.x >> 5;
  const int lane = threadIdx.x & 31;
  const int laneHi = lane >> 4, laneLo = lane & 15;
  const int pair = blockIdx.x * WAVES + w;
  if (pair >= PPAIRS) return;
  const int row0 = pair * 32;
  if (row0 >= meta[M_OFF + NREL]) return;  // beyond padded total
  // wave-uniform relation (buckets 32-padded -> pairs never straddle)
  int r = 0;
  if (row0 >= meta[M_OFF + 1]) r = 1;
  if (row0 >= meta[M_OFF + 2]) r = 2;
  if (row0 >= meta[M_OFF + 3]) r = 3;

  // gather A operands + per-row dst for both 16-edge subtiles
  v16h A0[2][4];
  int dstv[2][8];
#pragma unroll
  for (int s = 0; s < 2; ++s) {
    const int rs = row0 + s * 16;
    const int myPerm = perm[rs + laneLo];
    const int srow = (myPerm >= 0) ? src[myPerm] : 0;
    const float* rp = nf + (size_t)srow * DIMD;
#pragma unroll
    for (int c = 0; c < 4; ++c) A0[s][c] = load_a_f32(rp + c * 32 + 8 * laneHi);
#pragma unroll
    for (int v = 0; v < 8; ++v) {
      int p = perm[rs + 8 * laneHi + v];
      dstv[s][v] = (p >= 0) ? dst[p] : -1;
    }
  }

  const _Float16* Wb1 = B1s + (size_t)r * (4 * 4 * 512);
  const _Float16* Wb2 = B2s + (size_t)r * (2 * 4 * 512);
  const _Float16* Wb3 = B3s + (size_t)r * (2 * 8 * 512);

  // layer 1: [32,128] @ [128,64]  (32 WMMA, B loaded once per j)
#pragma unroll
  for (int j = 0; j < 4; ++j) {
    v16h bt[4];
#pragma unroll
    for (int kc = 0; kc < 4; ++kc) bt[kc] = load_b_tile(Wb1, kc * 4 + j, lane);
    const float bv = b1[r * HID + j * 16 + laneLo];
#pragma unroll
    for (int s = 0; s < 2; ++s) {
      v8f c = splat8(bv);
#pragma unroll
      for (int kc = 0; kc < 4; ++kc) c = wmma_f16(A0[s][kc], bt[kc], c);
      store_c_relu_lds(hbuf[w][s], c, j * 16, laneLo, laneHi);
    }
  }
  v16h a2[2][2];
#pragma unroll
  for (int s = 0; s < 2; ++s)
#pragma unroll
    for (int cc = 0; cc < 2; ++cc)
      a2[s][cc] = load_a_lds(hbuf[w][s], cc, laneLo, laneHi);

  // layer 2: [32,64] @ [64,64]  (16 WMMA)
#pragma unroll
  for (int j = 0; j < 4; ++j) {
    v16h bt[2];
#pragma unroll
    for (int kc = 0; kc < 2; ++kc) bt[kc] = load_b_tile(Wb2, kc * 4 + j, lane);
    const float bv = b2[r * HID + j * 16 + laneLo];
#pragma unroll
    for (int s = 0; s < 2; ++s) {
      v8f c = splat8(bv);
#pragma unroll
      for (int kc = 0; kc < 2; ++kc) c = wmma_f16(a2[s][kc], bt[kc], c);
      store_c_relu_lds(hbuf[w][s], c, j * 16, laneLo, laneHi);
    }
  }
  v16h a3[2][2];
#pragma unroll
  for (int s = 0; s < 2; ++s)
#pragma unroll
    for (int cc = 0; cc < 2; ++cc)
      a3[s][cc] = load_a_lds(hbuf[w][s], cc, laneLo, laneHi);

  // layer 3: [32,64] @ [64,128]  (32 WMMA) + relu + scatter-add
#pragma unroll
  for (int j = 0; j < 8; ++j) {
    v16h bt[2];
#pragma unroll
    for (int kc = 0; kc < 2; ++kc) bt[kc] = load_b_tile(Wb3, kc * 8 + j, lane);
    const float bv = b3[r * DIMD + j * 16 + laneLo];
#pragma unroll
    for (int s = 0; s < 2; ++s) {
      v8f c = splat8(bv);
#pragma unroll
      for (int kc = 0; kc < 2; ++kc) c = wmma_f16(a3[s][kc], bt[kc], c);
#pragma unroll
      for (int v = 0; v < 8; ++v) {
        float val = fmaxf(c[v], 0.f);
        if (dstv[s][v] >= 0 && val != 0.f) {
          size_t addr = (size_t)dstv[s][v] * (NREL * DIMD) + (size_t)r * DIMD +
                        j * 16 + laneLo;
          unsafeAtomicAdd(&agg[addr], val);
        }
      }
    }
  }
}

// ---------------- Node update kernel ----------------

__global__ void __launch_bounds__(256) node_update_kernel(
    const float* __restrict__ nf, const int* __restrict__ ntype,
    const float* __restrict__ hasin, const float* __restrict__ agg,
    const float* __restrict__ bu1, const float* __restrict__ bu2,
    const float* __restrict__ bu3,
    const _Float16* __restrict__ Bu1s, const _Float16* __restrict__ Bu2s,
    const _Float16* __restrict__ Bu3s,
    float* __restrict__ out) {
  __shared__ _Float16 hbuf[WAVES][16 * HID];
  const int w = threadIdx.x >> 5;
  const int lane = threadIdx.x & 31;
  const int laneHi = lane >> 4, laneLo = lane & 15;
  const int tile = blockIdx.x * WAVES + w;
  if (tile >= N_NODES / 16) return;
  const int n0 = tile * 16;
  _Float16* hb = hbuf[w];

  const int myRow = n0 + laneLo;
  const float* rp = nf + (size_t)myRow * DIMD;
  const float* ap = agg + (size_t)myRow * (NREL * DIMD);
  const float hi = hasin[myRow];

  // enc A operand: 20 K-chunks of 32 (4 from relu(x)*has_in, 16 from agg)
  v16h A[20];
#pragma unroll
  for (int c = 0; c < 4; ++c)
    A[c] = load_a_f32_relu_scale(rp + c * 32 + 8 * laneHi, hi);
#pragma unroll
  for (int c = 0; c < 16; ++c)
    A[4 + c] = load_a_f32(ap + c * 32 + 8 * laneHi);

  int ntv[8];
#pragma unroll
  for (int v = 0; v < 8; ++v) ntv[v] = ntype[n0 + 8 * laneHi + v];

  // layer 1: [16,640] @ [640,64]  (80 WMMA) — B tiles preloaded in groups of 4
#pragma unroll
  for (int j = 0; j < 4; ++j) {
    v8f c = splat8(bu1[j * 16 + laneLo]);
#pragma unroll
    for (int g = 0; g < 5; ++g) {
      v16h bt[4];
#pragma unroll
      for (int u = 0; u < 4; ++u)
        bt[u] = load_b_tile(Bu1s, (g * 4 + u) * 4 + j, lane);
#pragma unroll
      for (int u = 0; u < 4; ++u) c = wmma_f16(A[g * 4 + u], bt[u], c);
    }
    store_c_relu_lds(hb, c, j * 16, laneLo, laneHi);
  }
  v16h a2[2];
#pragma unroll
  for (int cc = 0; cc < 2; ++cc) a2[cc] = load_a_lds(hb, cc, laneLo, laneHi);

  // layer 2: [16,64] @ [64,64]  (8 WMMA)
#pragma unroll
  for (int j = 0; j < 4; ++j) {
    v16h bt[2];
#pragma unroll
    for (int kc = 0; kc < 2; ++kc) bt[kc] = load_b_tile(Bu2s, kc * 4 + j, lane);
    v8f c = splat8(bu2[j * 16 + laneLo]);
#pragma unroll
    for (int kc = 0; kc < 2; ++kc) c = wmma_f16(a2[kc], bt[kc], c);
    store_c_relu_lds(hb, c, j * 16, laneLo, laneHi);
  }
  v16h a3[2];
#pragma unroll
  for (int cc = 0; cc < 2; ++cc) a3[cc] = load_a_lds(hb, cc, laneLo, laneHi);

  // layer 3: [16,64] @ [64,128]  (16 WMMA) + fused masked writeback
#pragma unroll
  for (int j = 0; j < 8; ++j) {
    v16h bt[2];
#pragma unroll
    for (int kc = 0; kc < 2; ++kc) bt[kc] = load_b_tile(Bu3s, kc * 8 + j, lane);
    v8f c = splat8(bu3[j * 16 + laneLo]);
#pragma unroll
    for (int kc = 0; kc < 2; ++kc) c = wmma_f16(a3[kc], bt[kc], c);
#pragma unroll
    for (int v = 0; v < 8; ++v) {
      int rowv = n0 + v + 8 * laneHi;
      float val = (ntv[v] <= 1) ? c[v]
                                : nf[(size_t)rowv * DIMD + j * 16 + laneLo];
      out[(size_t)rowv * DIMD + j * 16 + laneLo] = val;
    }
  }
}

// ---------------- Host launcher ----------------

extern "C" void kernel_launch(void* const* d_in, const int* in_sizes, int n_in,
                              void* d_out, int out_size, void* d_ws, size_t ws_size,
                              hipStream_t stream) {
  const float* nf = (const float*)d_in[0];
  const int* src = (const int*)d_in[1];
  const int* dst = (const int*)d_in[2];
  const int* et = (const int*)d_in[3];
  const int* ntype = (const int*)d_in[4];
  const float* W1 = (const float*)d_in[5];
  const float* b1 = (const float*)d_in[6];
  const float* W2 = (const float*)d_in[7];
  const float* b2 = (const float*)d_in[8];
  const float* W3 = (const float*)d_in[9];
  const float* b3 = (const float*)d_in[10];
  const float* Wu1 = (const float*)d_in[11];
  const float* bu1 = (const float*)d_in[12];
  const float* Wu2 = (const float*)d_in[13];
  const float* bu2 = (const float*)d_in[14];
  const float* Wu3 = (const float*)d_in[15];
  const float* bu3 = (const float*)d_in[16];
  float* out = (float*)d_out;

  // workspace layout:
  // agg[N,512] f32 | hasin[N] f32 | swizzled f16 weights | meta[16] i32 | perm i32
  float* agg = (float*)d_ws;
  float* hasin = agg + (size_t)N_NODES * NREL * DIMD;
  _Float16* wsh = (_Float16*)(hasin + N_NODES);
  _Float16* B1s = wsh;                        // R * 16 tiles * 512
  _Float16* B2s = B1s + (size_t)NREL * 8192;  // R * 8 tiles
  _Float16* B3s = B2s + (size_t)NREL * 4096;  // R * 16 tiles
  _Float16* Bu1s = B3s + (size_t)NREL * 8192; // 80 tiles
  _Float16* Bu2s = Bu1s + 40960;              // 8 tiles
  _Float16* Bu3s = Bu2s + 4096;               // 16 tiles
  int* meta = (int*)(Bu3s + 8192);            // 16 ints
  int* perm = meta + 16;                      // PERM_N ints

  // zero agg + hasin; zero meta; fill perm with -1 sentinels
  hipMemsetAsync(d_ws, 0,
                 (size_t)N_NODES * NREL * DIMD * sizeof(float) +
                     (size_t)N_NODES * sizeof(float),
                 stream);
  hipMemsetAsync(meta, 0, 16 * sizeof(int), stream);
  hipMemsetAsync(perm, 0xFF, (size_t)PERM_N * sizeof(int), stream);

  for (int r = 0; r < NREL; ++r) {
    swizzle_w_kernel<<<(8192 + 255) / 256, 256, 0, stream>>>(
        W1 + (size_t)r * 8192, B1s + (size_t)r * 8192, 128, 64);
    swizzle_w_kernel<<<(4096 + 255) / 256, 256, 0, stream>>>(
        W2 + (size_t)r * 4096, B2s + (size_t)r * 4096, 64, 64);
    swizzle_w_kernel<<<(8192 + 255) / 256, 256, 0, stream>>>(
        W3 + (size_t)r * 8192, B3s + (size_t)r * 8192, 64, 128);
  }
  swizzle_w_kernel<<<(40960 + 255) / 256, 256, 0, stream>>>(Wu1, Bu1s, 640, 64);
  swizzle_w_kernel<<<(4096 + 255) / 256, 256, 0, stream>>>(Wu2, Bu2s, 64, 64);
  swizzle_w_kernel<<<(8192 + 255) / 256, 256, 0, stream>>>(Wu3, Bu3s, 64, 128);

  // relation counting-sort: histogram -> scan -> placement
  hist_kernel<<<(N_EDGES + 255) / 256, 256, 0, stream>>>(et, meta, N_EDGES);
  scan_kernel<<<1, 1, 0, stream>>>(meta);
  place_kernel<<<(N_EDGES + 255) / 256, 256, 0, stream>>>(et, meta, perm, N_EDGES);

  hasin_kernel<<<(N_EDGES + 255) / 256, 256, 0, stream>>>(dst, hasin, N_EDGES);

  edge_msg_kernel<<<(PPAIRS + WAVES - 1) / WAVES, 256, 0, stream>>>(
      nf, src, dst, meta, perm, b1, b2, b3, B1s, B2s, B3s, agg);

  node_update_kernel<<<(N_NODES / 16 + WAVES - 1) / WAVES, 256, 0, stream>>>(
      nf, ntype, hasin, agg, bu1, bu2, bu3, Bu1s, Bu2s, Bu3s, out);
}
